// _NeighborAggregator_22686017258243
// MI455X (gfx1250) — compile-verified, matrix operand order checked by reading, not run on hardware
//
#include <hip/hip_runtime.h>
#include <hip/hip_bf16.h>
#include <math.h>

// ---------------------------------------------------------------------------
// Problem constants (from reference)
// ---------------------------------------------------------------------------
static constexpr int B_ = 16, T_ = 256, K_ = 32, IN_ = 128, H_ = 128;
static constexpr int R_ = B_ * T_;          // 4096 rows of (b,t)

typedef _Float16 v16h __attribute__((ext_vector_type(16)));
typedef _Float16 v8h  __attribute__((ext_vector_type(8)));
typedef float    v8f  __attribute__((ext_vector_type(8)));
typedef float    v4f  __attribute__((ext_vector_type(4)));

#define WMMA16(a, b, c) \
  __builtin_amdgcn_wmma_f32_16x16x32_f16(false, (a), false, (b), (short)0, (c), false, false)

__device__ __forceinline__ float gelu_exact(float x) {
  return 0.5f * x * (1.0f + erff(x * 0.70710678118654752f));
}

__device__ __forceinline__ v8h ld8_f32_as_f16(const float* __restrict__ p) {
  v4f x = *(const v4f*)p;
  v4f y = *(const v4f*)(p + 4);
  v8h r;
  r[0] = (_Float16)x[0]; r[1] = (_Float16)x[1]; r[2] = (_Float16)x[2]; r[3] = (_Float16)x[3];
  r[4] = (_Float16)y[0]; r[5] = (_Float16)y[1]; r[6] = (_Float16)y[2]; r[7] = (_Float16)y[3];
  return r;
}

__device__ __forceinline__ v16h cat8(v8h lo, v8h hi) {
  return __builtin_shufflevector(lo, hi, 0, 1, 2, 3, 4, 5, 6, 7, 8, 9, 10, 11, 12, 13, 14, 15);
}

// ---------------------------------------------------------------------------
// K0: pack an f32 weight [Kdim=32*S, 128] into per-lane WMMA B-fragment order.
// packed[((s*8+ni)*32+lane)*16 + j] = (f16)W[s*32 + patK(lane,j)][ni*16 + (lane&15)]
// patK(lane,j) = (lane>>4)*8 + (j&7) + (j>=8 ? 16 : 0)   (ISA 16-bit operand layout)
// ---------------------------------------------------------------------------
__global__ __launch_bounds__(256) void pack_weight_kernel(
    const float* __restrict__ W, _Float16* __restrict__ dst, int S) {
  int gid = blockIdx.x * 256 + threadIdx.x;      // enumerates (s, ni, lane)
  if (gid >= S * 256) return;
  int lane = gid & 31;
  int sni  = gid >> 5;
  int s    = sni >> 3;
  int ni   = sni & 7;
  int n    = ni * 16 + (lane & 15);
  int kbase = s * 32 + ((lane >> 4) * 8);
#pragma unroll
  for (int j = 0; j < 16; ++j) {
    int k = kbase + (j & 7) + ((j >> 3) * 16);
    dst[gid * 16 + j] = (_Float16)W[k * 128 + n];
  }
}

// ---------------------------------------------------------------------------
// K1: center_embed = center_seq @ Wc + bc   (4096x128 @ 128x128)
// One block = 16 rows; 8 waves, wave w owns N-tile w; K-loop of 4 WMMAs.
// ---------------------------------------------------------------------------
__global__ __launch_bounds__(256) void center_embed_kernel(
    const float* __restrict__ cs, const float* __restrict__ bc,
    const _Float16* __restrict__ WcP,
    float* __restrict__ cF32, _Float16* __restrict__ cF16) {
  const int tid = threadIdx.x, lane = tid & 31, wid = tid >> 5;
  const int hb = lane >> 4, lr = lane & 15;
  const int row0 = blockIdx.x * 16;
  const int col = wid * 16 + lr;

  v8f c;
  float bias = bc[col];
#pragma unroll
  for (int v = 0; v < 8; ++v) c[v] = bias;

  const float* arow = cs + (size_t)(row0 + lr) * IN_;
#pragma unroll
  for (int s = 0; s < 4; ++s) {
    int kk0 = s * 32 + hb * 8;
    v16h a = cat8(ld8_f32_as_f16(arow + kk0), ld8_f32_as_f16(arow + kk0 + 16));
    v16h bf = *(const v16h*)(WcP + ((size_t)(s * 8 + wid) * 32 + lane) * 16);
    c = WMMA16(a, bf, c);
  }
#pragma unroll
  for (int v = 0; v < 8; ++v) {
    int m = v + 8 * hb;
    size_t idx = (size_t)(row0 + m) * H_ + col;
    cF32[idx] = c[v];
    cF16[idx] = (_Float16)c[v];
  }
}

// ---------------------------------------------------------------------------
// K2: per-(b,t) block. neighbor GEMM -> score GEMM -> softmax -> aggregate.
// ---------------------------------------------------------------------------
__global__ __launch_bounds__(256) void neighbor_kernel(
    const float* __restrict__ nseq,   // [B,T,K,128]
    const float* __restrict__ nstat,  // [B,K,32]
    const float* __restrict__ edgef,  // [B,K,32]
    const int*   __restrict__ nmask,  // [B,K]
    const float* __restrict__ bn, const float* __restrict__ bs1,
    const float* __restrict__ Ws2, const float* __restrict__ bs2,
    const _Float16* __restrict__ WnP, const _Float16* __restrict__ Ws1P,
    const _Float16* __restrict__ cF16,
    _Float16* __restrict__ aggF16, float* __restrict__ outW) {
  const int bt = blockIdx.x;
  const int b = bt >> 8;               // T_ == 256
  const int tid = threadIdx.x, lane = tid & 31, wid = tid >> 5;
  const int hb = lane >> 4, lr = lane & 15;
  const int col = wid * 16 + lr;

  __shared__ _Float16 ne[32][136];     // neighbor_embed f16 (A operand + aggregation)
  __shared__ float    s1[32][136];     // gelu(score_in @ Ws1) f32 (for precise Ws2 dot)
  __shared__ _Float16 ctr[128];        // this row's center_embed f16
  __shared__ float    wgt[32];

  if (tid < 128) ctr[tid] = cF16[(size_t)bt * H_ + tid];

  // ---- stage 1: neighbor_embed = concat(seq,static,edge) @ Wn + bn  (32x192 @ 192x128)
  {
    float bias = bn[col];
#pragma unroll
    for (int mi = 0; mi < 2; ++mi) {
      v8f c;
#pragma unroll
      for (int v = 0; v < 8; ++v) c[v] = bias;
      int krow = mi * 16 + lr;
      const float* seqrow = nseq + ((size_t)bt * K_ + krow) * IN_;
      const float* strow  = nstat + ((size_t)b * K_ + krow) * 32;
      const float* edrow  = edgef + ((size_t)b * K_ + krow) * 32;
#pragma unroll
      for (int s = 0; s < 6; ++s) {
        int kk0 = s * 32 + hb * 8;     // both 8-chunks stay inside one source tensor
        v8h lo, hi;
        if (s < 4)      { lo = ld8_f32_as_f16(seqrow + kk0);      hi = ld8_f32_as_f16(seqrow + kk0 + 16); }
        else if (s == 4){ lo = ld8_f32_as_f16(strow + kk0 - 128); hi = ld8_f32_as_f16(strow + kk0 - 112); }
        else            { lo = ld8_f32_as_f16(edrow + kk0 - 160); hi = ld8_f32_as_f16(edrow + kk0 - 144); }
        v16h a  = cat8(lo, hi);
        v16h bf = *(const v16h*)(WnP + ((size_t)(s * 8 + wid) * 32 + lane) * 16);
        c = WMMA16(a, bf, c);
      }
#pragma unroll
      for (int v = 0; v < 8; ++v) ne[mi * 16 + v + 8 * hb][col] = (_Float16)c[v];
    }
  }
  __syncthreads();

  // ---- stage 2: S1 = gelu([center | neighbor_embed] @ Ws1 + bs1)  (32x256 @ 256x128)
  {
    float bias = bs1[col];
#pragma unroll
    for (int mi = 0; mi < 2; ++mi) {
      v8f c;
#pragma unroll
      for (int v = 0; v < 8; ++v) c[v] = bias;
      int krow = mi * 16 + lr;
#pragma unroll
      for (int s = 0; s < 8; ++s) {
        int kk0 = s * 32 + hb * 8;
        v8h lo, hi;
        if (s < 4) { lo = *(const v8h*)&ctr[kk0];            hi = *(const v8h*)&ctr[kk0 + 16]; }
        else       { lo = *(const v8h*)&ne[krow][kk0 - 128]; hi = *(const v8h*)&ne[krow][kk0 - 112]; }
        v16h a  = cat8(lo, hi);
        v16h bf = *(const v16h*)(Ws1P + ((size_t)(s * 8 + wid) * 32 + lane) * 16);
        c = WMMA16(a, bf, c);
      }
#pragma unroll
      for (int v = 0; v < 8; ++v) s1[mi * 16 + v + 8 * hb][col] = gelu_exact(c[v]);
    }
  }
  __syncthreads();

  // ---- stage 3+4: score = S1 @ Ws2 + bs2; masked softmax over K (wave 0, wave32)
  if (tid < 32) {
    int k = tid;
    float sum = bs2[0];
    for (int h = 0; h < 128; ++h) sum += s1[k][h] * Ws2[h];
    int mk = nmask[b * K_ + k];
    float score = mk ? sum : -INFINITY;
    int am = mk ? 1 : 0;
    float mx = score;
#pragma unroll
    for (int off = 16; off > 0; off >>= 1) {
      mx = fmaxf(mx, __shfl_xor(mx, off, 32));
      am |= __shfl_xor(am, off, 32);
    }
    float w = 0.0f;
    if (am) {
      float e = expf(score - mx);          // masked lanes: exp(-inf) = 0
      float se = e;
#pragma unroll
      for (int off = 16; off > 0; off >>= 1) se += __shfl_xor(se, off, 32);
      w = e / se;
    }
    wgt[k] = w;
    outW[(size_t)bt * K_ + k] = w;
  }
  __syncthreads();

  // ---- stage 5: aggregated[h] = sum_k w[k] * neighbor_embed[k][h]
  if (tid < 128) {
    float acc = 0.0f;
#pragma unroll 8
    for (int k = 0; k < 32; ++k) acc += wgt[k] * (float)ne[k][tid];
    aggF16[(size_t)bt * H_ + tid] = (_Float16)acc;
  }
}

// ---------------------------------------------------------------------------
// K3: out = gelu([center | aggregated] @ Wo1 + bo1) @ Wo2 + bo2 + center_embed
// One block = 16 (b,t) rows.
// ---------------------------------------------------------------------------
__global__ __launch_bounds__(256) void output_kernel(
    const float* __restrict__ bo1, const float* __restrict__ bo2,
    const _Float16* __restrict__ Wo1P, const _Float16* __restrict__ Wo2P,
    const float* __restrict__ cF32, const _Float16* __restrict__ cF16,
    const _Float16* __restrict__ aggF16, float* __restrict__ out) {
  const int tid = threadIdx.x, lane = tid & 31, wid = tid >> 5;
  const int hb = lane >> 4, lr = lane & 15;
  const int row0 = blockIdx.x * 16;
  const int col = wid * 16 + lr;

  __shared__ _Float16 s1h[16][136];

  // GEMM1: [center | agg] (16x256) @ Wo1 -> gelu -> LDS f16
  {
    v8f c;
    float bias = bo1[col];
#pragma unroll
    for (int v = 0; v < 8; ++v) c[v] = bias;
    const _Float16* crow = cF16 + (size_t)(row0 + lr) * H_;
    const _Float16* arow = aggF16 + (size_t)(row0 + lr) * H_;
#pragma unroll
    for (int s = 0; s < 8; ++s) {
      int kk0 = s * 32 + hb * 8;
      v8h lo, hi;
      if (s < 4) { lo = *(const v8h*)(crow + kk0);       hi = *(const v8h*)(crow + kk0 + 16); }
      else       { lo = *(const v8h*)(arow + kk0 - 128); hi = *(const v8h*)(arow + kk0 - 112); }
      v16h a  = cat8(lo, hi);
      v16h bf = *(const v16h*)(Wo1P + ((size_t)(s * 8 + wid) * 32 + lane) * 16);
      c = WMMA16(a, bf, c);
    }
#pragma unroll
    for (int v = 0; v < 8; ++v) s1h[v + 8 * hb][col] = (_Float16)gelu_exact(c[v]);
  }
  __syncthreads();

  // GEMM2: (16x128) @ Wo2 + bo2 + residual
  {
    v8f c;
    float bias = bo2[col];
#pragma unroll
    for (int v = 0; v < 8; ++v) c[v] = bias;
#pragma unroll
    for (int s = 0; s < 4; ++s) {
      int kk0 = s * 32 + hb * 8;
      v8h lo = *(const v8h*)&s1h[lr][kk0];
      v8h hi = *(const v8h*)&s1h[lr][kk0 + 16];
      v16h a  = cat8(lo, hi);
      v16h bf = *(const v16h*)(Wo2P + ((size_t)(s * 8 + wid) * 32 + lane) * 16);
      c = WMMA16(a, bf, c);
    }
#pragma unroll
    for (int v = 0; v < 8; ++v) {
      size_t idx = (size_t)(row0 + v + 8 * hb) * H_ + col;
      out[idx] = c[v] + cF32[idx];
    }
  }
}

// ---------------------------------------------------------------------------
// Launcher
// ---------------------------------------------------------------------------
extern "C" void kernel_launch(void* const* d_in, const int* in_sizes, int n_in,
                              void* d_out, int out_size, void* d_ws, size_t ws_size,
                              hipStream_t stream) {
  const float* center_seq  = (const float*)d_in[0];
  const float* neighbor_seq = (const float*)d_in[1];
  const float* neighbor_static = (const float*)d_in[2];
  const float* edge_features = (const float*)d_in[3];
  const int*   neighbor_mask = (const int*)d_in[4];
  const float* Wc  = (const float*)d_in[5];
  const float* bc  = (const float*)d_in[6];
  const float* Wn  = (const float*)d_in[7];
  const float* bn  = (const float*)d_in[8];
  const float* Ws1 = (const float*)d_in[9];
  const float* bs1 = (const float*)d_in[10];
  const float* Ws2 = (const float*)d_in[11];
  const float* bs2 = (const float*)d_in[12];
  const float* Wo1 = (const float*)d_in[13];
  const float* bo1 = (const float*)d_in[14];
  const float* Wo2 = (const float*)d_in[15];
  const float* bo2 = (const float*)d_in[16];

  float* outMain = (float*)d_out;                 // [B,T,H]
  float* outW    = outMain + (size_t)R_ * H_;     // [B,T,K]

  // workspace layout (bytes)
  char* ws = (char*)d_ws;
  _Float16* WcP   = (_Float16*)(ws + 0);          //  32768
  _Float16* WnP   = (_Float16*)(ws + 32768);      //  49152
  _Float16* Ws1P  = (_Float16*)(ws + 81920);      //  65536
  _Float16* Wo1P  = (_Float16*)(ws + 147456);     //  65536
  _Float16* Wo2P  = (_Float16*)(ws + 212992);     //  32768
  float*    cF32  = (float*)   (ws + 245760);     // 2097152
  _Float16* cF16  = (_Float16*)(ws + 2342912);    // 1048576
  _Float16* aggF16= (_Float16*)(ws + 3391488);    // 1048576  (total 4440064 B)

  // stage 0: pack weights into WMMA fragment layout
  pack_weight_kernel<<<4, 256, 0, stream>>>(Wc,  WcP,  4);
  pack_weight_kernel<<<6, 256, 0, stream>>>(Wn,  WnP,  6);
  pack_weight_kernel<<<8, 256, 0, stream>>>(Ws1, Ws1P, 8);
  pack_weight_kernel<<<8, 256, 0, stream>>>(Wo1, Wo1P, 8);
  pack_weight_kernel<<<4, 256, 0, stream>>>(Wo2, Wo2P, 4);

  // stage 1: center embedding
  center_embed_kernel<<<R_ / 16, 256, 0, stream>>>(center_seq, bc, WcP, cF32, cF16);

  // stage 2: neighbor embed + attention + aggregation (one block per (b,t))
  neighbor_kernel<<<R_, 256, 0, stream>>>(
      neighbor_seq, neighbor_static, edge_features, neighbor_mask,
      bn, bs1, Ws2, bs2, WnP, Ws1P, cF16, aggF16, outW);

  // stage 3: output MLP + residual
  output_kernel<<<R_ / 16, 256, 0, stream>>>(bo1, bo2, Wo1P, Wo2P, cF32, cF16, aggF16, outMain);
}
